// RBF_84112639525231
// MI455X (gfx1250) — matrix operand order repaired and checked by reading, not compiled, not run on hardware
//
#include <hip/hip_runtime.h>
#include <hip/hip_bf16.h>

typedef __bf16 bf16;
typedef __attribute__((ext_vector_type(4)))  bf16  v4bf;
typedef __attribute__((ext_vector_type(8)))  bf16  v8bf;
typedef __attribute__((ext_vector_type(16))) bf16  v16bf;
typedef __attribute__((ext_vector_type(4)))  float v4f;
typedef __attribute__((ext_vector_type(8)))  float v8f;

#define NI 4096   // inputs rows (columns of hidden)
#define DI 512    // feature dim
#define HI 4096   // hidden count (rows of hidden)
#define OI 4096   // output dim

// GEMM tiling: 256x128 block, 8 waves of 64x64, K-step 32
#define BM 256
#define BN 128
#define BK 32
#define LDT 40                      // padded LDS row stride (bf16 elems) -> 80B, conflict-free
#define A_LDS (BM * LDT)            // 10240 elems
#define B_LDS (BN * LDT)            // 5120 elems

// ---------------------------------------------------------------------------
// CDNA5 async global->LDS copy (ASYNCcnt-tracked), bypasses VGPR staging.
// LDS byte address = low 32 bits of generic pointer (LDS aperture mapping).
// ---------------------------------------------------------------------------
__device__ __forceinline__ void async_ld_b128(const bf16* g, bf16* l) {
    asm volatile("global_load_async_to_lds_b128 %0, %1, off"
                 :: "v"((uint32_t)(size_t)l), "v"((uint64_t)(size_t)g)
                 : "memory");
}
__device__ __forceinline__ void wait_async0() {
#if __has_builtin(__builtin_amdgcn_s_wait_asynccnt)
    __builtin_amdgcn_s_wait_asynccnt(0);
#else
    asm volatile("s_wait_asynccnt 0x0" ::: "memory");
#endif
}

// ---------------------------------------------------------------------------
// Row-wise squared-norm + f32->bf16 conversion (D = 512, 128 threads/row)
// ---------------------------------------------------------------------------
__global__ __launch_bounds__(128)
void row_norms_convert(const float* __restrict__ src, float* __restrict__ norms,
                       bf16* __restrict__ dst) {
    const int row = blockIdx.x;
    const int t   = threadIdx.x;
    const v4f v = ((const v4f*)(src + (size_t)row * DI))[t];
    float s = v[0]*v[0] + v[1]*v[1] + v[2]*v[2] + v[3]*v[3];
    v4bf o;
    o[0] = (bf16)v[0]; o[1] = (bf16)v[1]; o[2] = (bf16)v[2]; o[3] = (bf16)v[3];
    ((v4bf*)(dst + (size_t)row * DI))[t] = o;
    #pragma unroll
    for (int off = 16; off; off >>= 1) s += __shfl_xor(s, off, 32);
    __shared__ float part[4];
    if ((t & 31) == 0) part[t >> 5] = s;
    __syncthreads();
    if (t == 0) norms[row] = part[0] + part[1] + part[2] + part[3];
}

// ---------------------------------------------------------------------------
// W(k,j) f32 -> Wt(j,k) bf16, 32x32 LDS-tiled transpose
// ---------------------------------------------------------------------------
__global__ __launch_bounds__(256)
void transpose_convert(const float* __restrict__ W, bf16* __restrict__ Wt) {
    __shared__ float tile[32][33];
    const int j0 = blockIdx.x * 32, k0 = blockIdx.y * 32;
    const int tx = threadIdx.x, ty = threadIdx.y;   // (32, 8)
    #pragma unroll
    for (int q = 0; q < 4; ++q)
        tile[ty + 8*q][tx] = W[(size_t)(k0 + ty + 8*q) * HI + j0 + tx];
    __syncthreads();
    #pragma unroll
    for (int q = 0; q < 4; ++q)
        Wt[(size_t)(j0 + ty + 8*q) * OI + k0 + tx] = (bf16)tile[tx][ty + 8*q];
}

// ---------------------------------------------------------------------------
// Deterministic column-sum reduction over the 16 per-block partials
// ---------------------------------------------------------------------------
__global__ __launch_bounds__(256)
void reduce_colsum(const float* __restrict__ P, float* __restrict__ colsum) {
    const int n = blockIdx.x * 256 + threadIdx.x;
    float s = 0.f;
    #pragma unroll
    for (int b = 0; b < HI / BM; ++b) s += P[(size_t)b * NI + n];
    colsum[n] = s;
}

// ---------------------------------------------------------------------------
// hidden[i,k] /= colsum[k]  (bf16 in place, 8 elems/thread)
// ---------------------------------------------------------------------------
__global__ __launch_bounds__(256)
void normalize_hidden(bf16* __restrict__ h, const float* __restrict__ colsum) {
    const size_t base = ((size_t)blockIdx.x * 256 + threadIdx.x) * 8;
    v8bf v = *(const v8bf*)(h + base);
    const int k0 = (int)(base & (NI - 1));
    v8bf o;
    #pragma unroll
    for (int e = 0; e < 8; ++e)
        o[e] = (bf16)((float)v[e] / colsum[k0 + e]);
    *(v8bf*)(h + base) = o;
}

// ---------------------------------------------------------------------------
// bf16 WMMA GEMM skeleton.  A: (M,K) row-major, B: (Ncols,K) "row-major by n"
// (i.e. each B row n holds K contiguous contraction values).
// EPI==0: out[m,n] = acc + bias[n]         (f32)
// EPI==1: h = exp(-0.5*(c2[m]-2*acc+x2[n])) -> bf16 store + column partials
// ---------------------------------------------------------------------------
template <int EPI>
__global__ __launch_bounds__(256)
void gemm_bf16(const bf16* __restrict__ A, const bf16* __restrict__ B, int K,
               float* __restrict__ out, const float* __restrict__ bias,
               bf16* __restrict__ hOut, const float* __restrict__ c2,
               const float* __restrict__ x2, float* __restrict__ colPart) {
    __shared__ bf16 smem[2][A_LDS + B_LDS];

    const int tid  = threadIdx.x;
    const int lane = tid & 31;
    const int wave = tid >> 5;
    const int wm = wave >> 1;        // 0..3
    const int wn = wave & 1;         // 0..1
    const int lh = lane >> 4;        // 0/1
    const int lm = lane & 15;

    const int m0 = blockIdx.y * BM;
    const int n0 = blockIdx.x * BN;

    // async global -> LDS tile copy: 4 A chunks + 2 B chunks of 16B per thread
    auto issue_tile = [&](int buf, int kt) {
        const bf16* Ag = A + (size_t)m0 * K + kt * BK;
        const bf16* Bg = B + (size_t)n0 * K + kt * BK;
        bf16* As = smem[buf];
        bf16* Bs = smem[buf] + A_LDS;
        #pragma unroll
        for (int q = 0; q < 4; ++q) {
            int idx = q * 256 + tid, r = idx >> 2, c = idx & 3;
            async_ld_b128(Ag + (size_t)r * K + c * 8, As + r * LDT + c * 8);
        }
        #pragma unroll
        for (int q = 0; q < 2; ++q) {
            int idx = q * 256 + tid, r = idx >> 2, c = idx & 3;
            async_ld_b128(Bg + (size_t)r * K + c * 8, Bs + r * LDT + c * 8);
        }
    };
    // A fragment: lane holds row m=lm, K in {8*lh..8*lh+7} u {16+8*lh..+7}
    auto afrag = [&](const bf16* As, int am) -> v16bf {
        const bf16* p = As + (wm * 64 + am * 16 + lm) * LDT + lh * 8;
        v8bf lo = *(const v8bf*)p;
        v8bf hi = *(const v8bf*)(p + 16);
        return __builtin_shufflevector(lo, hi, 0,1,2,3,4,5,6,7,8,9,10,11,12,13,14,15);
    };
    // B fragment: lane holds col n=lm, K in {8*lh..8*lh+7} u {16+8*lh..+7}
    auto bfrag = [&](const bf16* Bs, int bn) -> v16bf {
        const bf16* p = Bs + (wn * 64 + bn * 16 + lm) * LDT + lh * 8;
        v8bf lo = *(const v8bf*)p;
        v8bf hi = *(const v8bf*)(p + 16);
        return __builtin_shufflevector(lo, hi, 0,1,2,3,4,5,6,7,8,9,10,11,12,13,14,15);
    };

    v8f acc[4][4];
    #pragma unroll
    for (int i = 0; i < 4; ++i)
        #pragma unroll
        for (int j = 0; j < 4; ++j)
            #pragma unroll
            for (int e = 0; e < 8; ++e) acc[i][j][e] = 0.f;

    const int KT = K / BK;
    issue_tile(0, 0);
    wait_async0();
    __syncthreads();

    for (int kt = 0; kt < KT; ++kt) {
        const int cur = kt & 1;
        // prefetch next K-tile into the other buffer; safe: the barrier that
        // ended iteration kt-1 ordered all reads of that buffer before now.
        if (kt + 1 < KT) issue_tile(cur ^ 1, kt + 1);
        const bf16* As = smem[cur];
        const bf16* Bs = smem[cur] + A_LDS;
        v16bf af[4], bv[4];
        #pragma unroll
        for (int am = 0; am < 4; ++am) af[am] = afrag(As, am);
        #pragma unroll
        for (int bn = 0; bn < 4; ++bn) bv[bn] = bfrag(Bs, bn);
        #pragma unroll
        for (int am = 0; am < 4; ++am)
            #pragma unroll
            for (int bn = 0; bn < 4; ++bn)
                acc[am][bn] = __builtin_amdgcn_wmma_f32_16x16x32_bf16(
                    false, af[am], false, bv[bn], (short)0, acc[am][bn], false, false);
        if (kt + 1 < KT) {
            wait_async0();      // my async writes into cur^1 have landed
            __syncthreads();    // everyone's writes visible, all reads of cur done
        }
    }

    // C/D layout: lane: n = lm; VGPR r: m = r + 8*lh
    if constexpr (EPI == 0) {
        #pragma unroll
        for (int bn = 0; bn < 4; ++bn) {
            const int n = n0 + wn * 64 + bn * 16 + lm;
            const float bvv = bias[n];
            #pragma unroll
            for (int am = 0; am < 4; ++am) {
                const int mb = m0 + wm * 64 + am * 16 + lh * 8;
                #pragma unroll
                for (int r = 0; r < 8; ++r)
                    out[(size_t)(mb + r) * NI + n] = acc[am][bn][r] + bvv;
            }
        }
    } else {
        __shared__ float colpart[4][BN];
        #pragma unroll
        for (int bn = 0; bn < 4; ++bn) {
            const int n = n0 + wn * 64 + bn * 16 + lm;
            const float xv = x2[n];
            float cs = 0.f;
            #pragma unroll
            for (int am = 0; am < 4; ++am) {
                const int mb = m0 + wm * 64 + am * 16 + lh * 8;
                #pragma unroll
                for (int r = 0; r < 8; ++r) {
                    float d2 = c2[mb + r] - 2.0f * acc[am][bn][r] + xv;
                    bf16 hb = (bf16)__expf(-0.5f * d2);
                    hOut[(size_t)(mb + r) * NI + n] = hb;
                    cs += (float)hb;       // sum the rounded values (consistent norm)
                }
            }
            cs += __shfl_xor(cs, 16, 32);  // fold lane-halves (same column)
            if (lh == 0) colpart[wm][wn * 64 + bn * 16 + lm] = cs;
        }
        __syncthreads();
        if (tid < BN)                       // deterministic 4-way wave fold
            colPart[(size_t)blockIdx.y * NI + n0 + tid] =
                colpart[0][tid] + colpart[1][tid] + colpart[2][tid] + colpart[3][tid];
    }
}

// ---------------------------------------------------------------------------
extern "C" void kernel_launch(void* const* d_in, const int* in_sizes, int n_in,
                              void* d_out, int out_size, void* d_ws, size_t ws_size,
                              hipStream_t stream) {
    const float* inputs = (const float*)d_in[0];
    const float* hw     = (const float*)d_in[1];
    const float* ow     = (const float*)d_in[2];
    const float* ob     = (const float*)d_in[3];
    float* out = (float*)d_out;

    char* ws = (char*)d_ws;
    size_t off = 0;
    auto alloc = [&](size_t bytes) -> void* {
        void* p = ws + off;
        off += (bytes + 255) & ~(size_t)255;
        return p;
    };
    float* x2     = (float*)alloc((size_t)NI * 4);
    float* c2     = (float*)alloc((size_t)HI * 4);
    float* colsum = (float*)alloc((size_t)NI * 4);
    float* P      = (float*)alloc((size_t)(HI / BM) * NI * 4);  // per-block col partials
    bf16*  Xb     = (bf16*) alloc((size_t)NI * DI * 2);
    bf16*  Cb     = (bf16*) alloc((size_t)HI * DI * 2);
    bf16*  Wt     = (bf16*) alloc((size_t)OI * HI * 2);
    bf16*  Hb     = (bf16*) alloc((size_t)HI * NI * 2);

    row_norms_convert<<<NI, 128, 0, stream>>>(inputs, x2, Xb);
    row_norms_convert<<<HI, 128, 0, stream>>>(hw, c2, Cb);
    transpose_convert<<<dim3(HI / 32, OI / 32), dim3(32, 8), 0, stream>>>(ow, Wt);

    // cross + RBF epilogue: A = hidden_weights(bf16), B rows = input rows, K = D
    gemm_bf16<1><<<dim3(NI / BN, HI / BM), 256, 0, stream>>>(
        Cb, Xb, DI, nullptr, nullptr, Hb, c2, x2, P);

    reduce_colsum<<<NI / 256, 256, 0, stream>>>(P, colsum);
    normalize_hidden<<<(int)(((size_t)HI * NI / 8) / 256), 256, 0, stream>>>(Hb, colsum);

    // final addmm: A = normalized hidden (bf16), B rows = Wt rows (j,k), K = N
    gemm_bf16<0><<<dim3(HI / BN, HI / BM), 256, 0, stream>>>(
        Hb, Wt, NI, out, ob, nullptr, nullptr, nullptr, nullptr);
}
